// RGCN_2637109920454
// MI455X (gfx1250) — matrix operand order repaired and checked by reading, not compiled, not run on hardware
//
#include <hip/hip_runtime.h>

// ---- problem constants (match reference) ----
#define NNODES  100000
#define NA      30000    // author rows (softmax output)
#define EDGES   1600000
#define RREL    6
#define NBASES  30

// ---- CDNA5 WMMA types ----
typedef __attribute__((ext_vector_type(16))) __bf16 v16bf;
typedef __attribute__((ext_vector_type(2)))  __bf16 v2bf;
typedef __attribute__((ext_vector_type(8)))  float  v8f;

#if defined(__has_builtin)
#if __has_builtin(__builtin_amdgcn_cvt_pk_bf16_f32)
#define HAVE_CVT_PK_BF16 1
#endif
#endif

__device__ __forceinline__ unsigned short f2bf_bits(float f) {
#ifdef HAVE_CVT_PK_BF16
  v2bf h = __builtin_amdgcn_cvt_pk_bf16_f32(f, f);
  union { __bf16 b; unsigned short s; } y; y.b = h[0]; return y.s;
#else
  union { float f; unsigned u; } x; x.f = f;
  unsigned r = x.u + 0x7FFFu + ((x.u >> 16) & 1u);   // round-to-nearest-even
  return (unsigned short)(r >> 16);
#endif
}

// ---------------------------------------------------------------------
// B pre-pack: logical B[k][n] (K x N) -> WMMA fragment order.
// Per 32x16 K-tile: lane = (((k%32)>>3)&1)<<4 | (n%16), element j = (k%8) | ((k%32)>=16)<<3.
// Each lane's 16 bf16 are CONTIGUOUS (32 B) -> two global_load_b128.
// ---------------------------------------------------------------------
__device__ __forceinline__ size_t bpack_idx(int k, int n, int tilesN) {
  int kt = k >> 5, kk = k & 31;
  int nt = n >> 4, col = n & 15;
  int lane = (((kk >> 3) & 1) << 4) | col;
  int j = (kk & 7) | ((kk >> 4) << 3);
  return ((((size_t)kt * tilesN + nt) << 9)) + (lane << 4) + j;
}

// ---- fragment loaders ----
__device__ __forceinline__ v16bf load_a_frag(const unsigned short* p) {
  // elements 0..7 = p[0..7], 8..15 = p[16..23]; both 16B-aligned chunks
  union { uint4 u[2]; v16bf v; } t;
  t.u[0] = *(const uint4*)p;
  t.u[1] = *(const uint4*)(p + 16);
  return t.v;
}
__device__ __forceinline__ v16bf load_a_frag(const float* p) {
  const float4 f0 = *(const float4*)(p);
  const float4 f1 = *(const float4*)(p + 4);
  const float4 f2 = *(const float4*)(p + 16);
  const float4 f3 = *(const float4*)(p + 20);
#ifdef HAVE_CVT_PK_BF16
  union { v2bf h[8]; v16bf v; } t;
  t.h[0] = __builtin_amdgcn_cvt_pk_bf16_f32(f0.x, f0.y);
  t.h[1] = __builtin_amdgcn_cvt_pk_bf16_f32(f0.z, f0.w);
  t.h[2] = __builtin_amdgcn_cvt_pk_bf16_f32(f1.x, f1.y);
  t.h[3] = __builtin_amdgcn_cvt_pk_bf16_f32(f1.z, f1.w);
  t.h[4] = __builtin_amdgcn_cvt_pk_bf16_f32(f2.x, f2.y);
  t.h[5] = __builtin_amdgcn_cvt_pk_bf16_f32(f2.z, f2.w);
  t.h[6] = __builtin_amdgcn_cvt_pk_bf16_f32(f3.x, f3.y);
  t.h[7] = __builtin_amdgcn_cvt_pk_bf16_f32(f3.z, f3.w);
  return t.v;
#else
  const float ff[16] = { f0.x, f0.y, f0.z, f0.w, f1.x, f1.y, f1.z, f1.w,
                         f2.x, f2.y, f2.z, f2.w, f3.x, f3.y, f3.z, f3.w };
  union { unsigned short s[16]; v16bf v; } t;
#pragma unroll
  for (int j = 0; j < 16; ++j) t.s[j] = f2bf_bits(ff[j]);
  return t.v;
#endif
}
__device__ __forceinline__ void store_c(float* p, float v)          { *p = v; }
__device__ __forceinline__ void store_c(unsigned short* p, float v) { *p = f2bf_bits(v); }

// =====================================================================
// WMMA GEMM: C[M,N] = A[M,K] * Bpacked[K,N] (+ bias[N])
//   AT: float (converted in-register) or unsigned short (bf16 bits)
//   OT: float or unsigned short (bf16 bits)
//   NT: N-tiles (16 cols each) per wave; one A-frag feeds NT WMMAs/k-step
//   M%16==0, K%32==0, N%(16*NT)==0  (all shapes here satisfy this)
// One wave per 16 x (16*NT) tile, 8 waves (256 thr) per block.
// =====================================================================
template <typename AT, typename OT, int NT, bool HAS_BIAS>
__global__ void gemm_wmma(const AT* __restrict__ A, int lda,
                          const unsigned short* __restrict__ Bp,
                          OT* __restrict__ C, int ldc,
                          const float* __restrict__ bias,
                          int M, int N, int K)
{
  const int wave = threadIdx.x >> 5;
  const int lane = threadIdx.x & 31;
  const int tilesN = N >> 4;
  const int grpN = tilesN / NT;
  const int totalTiles = (M >> 4) * grpN;
  const int tileIdx = blockIdx.x * 8 + wave;
  if (tileIdx >= totalTiles) return;     // wave-uniform exit (EXEC all-1 for WMMA)

  const int tm  = tileIdx / grpN;
  const int tg  = tileIdx - tm * grpN;
  const int row0 = tm << 4;
  const int tn0  = tg * NT;
  const int col  = lane & 15;
  const int arow = row0 + col;
  const int akoff = (lane < 16) ? 0 : 8;   // 16-bit A frag layout (ISA 7.12.2)

  v8f acc[NT];
#pragma unroll
  for (int t = 0; t < NT; ++t) {
    const float bi = HAS_BIAS ? bias[((tn0 + t) << 4) + col] : 0.0f;
#pragma unroll
    for (int p = 0; p < 8; ++p) acc[t][p] = bi;
  }

  for (int k0 = 0; k0 < K; k0 += 32) {
    const v16bf a = load_a_frag(A + (size_t)arow * lda + (k0 + akoff));
    const unsigned short* bbase =
        Bp + ((((size_t)(k0 >> 5)) * tilesN + tn0) << 9) + (lane << 4);
#pragma unroll
    for (int t = 0; t < NT; ++t) {
      const v16bf b = *(const v16bf*)(bbase + ((size_t)t << 9));
      acc[t] = __builtin_amdgcn_wmma_f32_16x16x32_bf16(false, a, false, b,
                                                       (short)0, acc[t], false, false);
    }
  }

  const int rbase = row0 + ((lane >> 4) << 3);   // D layout: VGPR p -> row p (+8 hi half)
#pragma unroll
  for (int t = 0; t < NT; ++t) {
    const int cn = ((tn0 + t) << 4) + col;
#pragma unroll
    for (int p = 0; p < 8; ++p)
      store_c(&C[(size_t)(rbase + p) * ldc + cn], acc[t][p]);
  }
}

// ---- weight prep (write directly in packed fragment order) ----
// B[k][n] = W[n*K+k]  (W is N x K row-major: lin_w out x in)
__global__ void k_transpose_pack(const float* __restrict__ W,
                                 unsigned short* __restrict__ Bt, int N, int K)
{
  int i = blockIdx.x * blockDim.x + threadIdx.x;
  if (i >= N * K) return;
  int k = i / N, n = i - k * N;
  Bt[bpack_idx(k, n, N >> 4)] = f2bf_bits(W[(size_t)n * K + k]);
}

// B[k][n] = W[k*N+n]  (root weights: already K x N)
__global__ void k_convert_pack(const float* __restrict__ W,
                               unsigned short* __restrict__ O, int N, int K)
{
  int i = blockIdx.x * blockDim.x + threadIdx.x;
  if (i >= N * K) return;
  int k = i / N, n = i - k * N;
  O[bpack_idx(k, n, N >> 4)] = f2bf_bits(W[i]);
}

// W[r][k][n] = sum_b comp[r,b]*bases[b,k,n], packed per relation (stride K*N)
__global__ void k_combine_bases_pack(const float* __restrict__ comp,
                                     const float* __restrict__ bases,
                                     unsigned short* __restrict__ Wout,
                                     int N, int K)
{
  int IO = K * N;
  int i = blockIdx.x * blockDim.x + threadIdx.x;
  if (i >= RREL * IO) return;
  int r = i / IO, io = i - r * IO;
  int k = io / N, n = io - k * N;
  float s = 0.0f;
#pragma unroll
  for (int b = 0; b < NBASES; ++b)
    s += comp[r * NBASES + b] * bases[(size_t)b * IO + io];
  Wout[(size_t)r * IO + bpack_idx(k, n, N >> 4)] = f2bf_bits(s);
}

// ---- graph ops ----
__global__ void k_count_edges(const int* __restrict__ et,
                              const int* __restrict__ dst,
                              int* __restrict__ cnt)
{
  int e = blockIdx.x * blockDim.x + threadIdx.x;
  if (e >= EDGES) return;
  atomicAdd(&cnt[(size_t)et[e] * NNODES + dst[e]], 1);
}

// H[dst,:] += xt_r[src,:] / max(cnt[r,dst],1) for edges of relation r.
// One thread per (edge, 4 channels): float4 gather + 4 f32 atomics.
// C is compile-time so edge/channel decomposition is shift/mask (no div).
template <int C>
__global__ void k_scatter_mean(const float* __restrict__ xt,
                               const int* __restrict__ src,
                               const int* __restrict__ dst,
                               const int* __restrict__ et,
                               const int* __restrict__ cnt,
                               int r, float* __restrict__ H)
{
  constexpr int CQ_SHIFT = (C == 64) ? 4 : 2;   // log2(C/4)
  constexpr int CQ_MASK  = (1 << CQ_SHIFT) - 1;
  int tid = blockIdx.x * blockDim.x + threadIdx.x;   // <= E * C/4 = 25.6M
  int e = tid >> CQ_SHIFT;
  if (e >= EDGES) return;
  if (et[e] != r) return;
  int q = tid & CQ_MASK;
  int d = dst[e];
  int n = cnt[r * NNODES + d];
  float inv = 1.0f / (float)(n > 1 ? n : 1);
  const float4 v = *(const float4*)(xt + (size_t)src[e] * C + (q << 2));
  float* hp = H + (size_t)d * C + (q << 2);
  atomicAdd(hp + 0, v.x * inv);
  atomicAdd(hp + 1, v.y * inv);
  atomicAdd(hp + 2, v.z * inv);
  atomicAdd(hp + 3, v.w * inv);
}

// row-wise softmax over 16 classes; one 16-lane half-wave per row
__global__ void k_softmax16(const float* __restrict__ H, float* __restrict__ out,
                            int rows)
{
  int wid  = (blockIdx.x * blockDim.x + threadIdx.x) >> 5;
  int lane = threadIdx.x & 31;
  int row  = wid * 2 + (lane >> 4);
  if (row >= rows) return;
  float v = H[(size_t)row * 16 + (lane & 15)];
  float m = v;
#pragma unroll
  for (int off = 8; off; off >>= 1) m = fmaxf(m, __shfl_xor(m, off, 32));
  float e = __expf(v - m);
  float s = e;
#pragma unroll
  for (int off = 8; off; off >>= 1) s += __shfl_xor(s, off, 32);
  out[(size_t)row * 16 + (lane & 15)] = e / s;
}

// =====================================================================
extern "C" void kernel_launch(void* const* d_in, const int* in_sizes, int n_in,
                              void* d_out, int out_size, void* d_ws, size_t ws_size,
                              hipStream_t stream)
{
  const float* x0     = (const float*)d_in[0];
  const float* x1     = (const float*)d_in[1];
  const float* x2     = (const float*)d_in[2];
  const float* lin_w0 = (const float*)d_in[3];
  const float* lin_b0 = (const float*)d_in[4];
  const float* lin_w1 = (const float*)d_in[5];
  const float* lin_b1 = (const float*)d_in[6];
  const float* lin_w2 = (const float*)d_in[7];
  const float* lin_b2 = (const float*)d_in[8];
  const float* bases1 = (const float*)d_in[9];
  const float* comp1  = (const float*)d_in[10];
  const float* root1  = (const float*)d_in[11];
  const float* bias1  = (const float*)d_in[12];
  const float* bases2 = (const float*)d_in[13];
  const float* comp2  = (const float*)d_in[14];
  const float* root2  = (const float*)d_in[15];
  const float* bias2  = (const float*)d_in[16];
  const int*   eidx   = (const int*)d_in[17];
  const int*   etype  = (const int*)d_in[18];
  const int*   esrc   = eidx;
  const int*   edst   = eidx + EDGES;
  float* out = (float*)d_out;

  // ---- workspace layout ----
  char* ws = (char*)d_ws;
  size_t off = 0;
  auto take = [&](size_t bytes) -> void* {
    void* p = ws + off;
    off += (bytes + 255) & ~(size_t)255;
    return p;
  };
  unsigned short* Xbf  = (unsigned short*)take((size_t)NNODES * 128 * 2);
  float*          H1   = (float*)take((size_t)NNODES * 64 * 4);
  float*          XT   = (float*)take((size_t)NNODES * 64 * 4);
  int*            CNT  = (int*)take((size_t)RREL * NNODES * 4);
  float*          H2   = (float*)take((size_t)NNODES * 16 * 4);
  unsigned short* WL0T = (unsigned short*)take(256 * 128 * 2);
  unsigned short* WL1T = (unsigned short*)take(512 * 128 * 2);
  unsigned short* WL2T = (unsigned short*)take(128 * 128 * 2);
  unsigned short* W1B  = (unsigned short*)take((size_t)RREL * 128 * 64 * 2);
  unsigned short* R1B  = (unsigned short*)take(128 * 64 * 2);
  unsigned short* W2B  = (unsigned short*)take((size_t)RREL * 64 * 16 * 2);
  unsigned short* R2B  = (unsigned short*)take(64 * 16 * 2);

  // ---- 1. weight prep (bf16, packed fragment order) ----
  k_transpose_pack<<<(256 * 128 + 255) / 256, 256, 0, stream>>>(lin_w0, WL0T, 128, 256);
  k_transpose_pack<<<(512 * 128 + 255) / 256, 256, 0, stream>>>(lin_w1, WL1T, 128, 512);
  k_transpose_pack<<<(128 * 128 + 255) / 256, 256, 0, stream>>>(lin_w2, WL2T, 128, 128);
  k_combine_bases_pack<<<(RREL * 128 * 64 + 255) / 256, 256, 0, stream>>>(comp1, bases1, W1B, 64, 128);
  k_combine_bases_pack<<<(RREL * 64 * 16 + 255) / 256, 256, 0, stream>>>(comp2, bases2, W2B, 16, 64);
  k_convert_pack<<<(128 * 64 + 255) / 256, 256, 0, stream>>>(root1, R1B, 64, 128);
  k_convert_pack<<<(64 * 16 + 255) / 256, 256, 0, stream>>>(root2, R2B, 16, 64);

  // tiles per grid: (M/16) * (N/16/NT), 8 waves per block
  auto gemmGrid = [](int M, int N, int NT) {
    return ((M >> 4) * ((N >> 4) / NT) + 7) / 8;
  };

  // ---- 2. input projections -> Xbf (bf16, +bias) ----
  gemm_wmma<float, unsigned short, 4, true><<<gemmGrid(NA, 128, 4), 256, 0, stream>>>(
      x0, 256, WL0T, Xbf, 128, lin_b0, NA, 128, 256);
  gemm_wmma<float, unsigned short, 4, true><<<gemmGrid(50000, 128, 4), 256, 0, stream>>>(
      x1, 512, WL1T, Xbf + (size_t)NA * 128, 128, lin_b1, 50000, 128, 512);
  gemm_wmma<float, unsigned short, 4, true><<<gemmGrid(20000, 128, 4), 256, 0, stream>>>(
      x2, 128, WL2T, Xbf + (size_t)80000 * 128, 128, lin_b2, 20000, 128, 128);

  // ---- 3. per-(relation,dst) edge counts (shared by both layers) ----
  hipMemsetAsync(CNT, 0, (size_t)RREL * NNODES * 4, stream);
  k_count_edges<<<(EDGES + 255) / 256, 256, 0, stream>>>(etype, edst, CNT);

  // ---- 4. RGCN layer 1: H1 = X@root1 + bias1, then += mean msgs per relation ----
  gemm_wmma<unsigned short, float, 4, true><<<gemmGrid(NNODES, 64, 4), 256, 0, stream>>>(
      Xbf, 128, R1B, H1, 64, bias1, NNODES, 64, 128);
  for (int r = 0; r < RREL; ++r) {
    gemm_wmma<unsigned short, float, 4, false><<<gemmGrid(NNODES, 64, 4), 256, 0, stream>>>(
        Xbf, 128, W1B + (size_t)r * 128 * 64, XT, 64, (const float*)nullptr, NNODES, 64, 128);
    k_scatter_mean<64><<<(EDGES * 16 + 255) / 256, 256, 0, stream>>>(
        XT, esrc, edst, etype, CNT, r, H1);
  }

  // ---- 5. RGCN layer 2: H2 = H1@root2 + bias2, then += mean msgs ----
  gemm_wmma<float, float, 1, true><<<gemmGrid(NNODES, 16, 1), 256, 0, stream>>>(
      H1, 64, R2B, H2, 16, bias2, NNODES, 16, 64);
  for (int r = 0; r < RREL; ++r) {
    gemm_wmma<float, float, 1, false><<<gemmGrid(NNODES, 16, 1), 256, 0, stream>>>(
        H1, 64, W2B + (size_t)r * 64 * 16, XT, 16, (const float*)nullptr, NNODES, 16, 64);
    k_scatter_mean<16><<<(EDGES * 4 + 255) / 256, 256, 0, stream>>>(
        XT, esrc, edst, etype, CNT, r, H2);
  }

  // ---- 6. softmax over first 30000 rows ----
  k_softmax16<<<(NA / 2 * 32 + 255) / 256, 256, 0, stream>>>(H2, out, NA);
}
